// QuantumLayer_45981919871255
// MI455X (gfx1250) — matrix operand order repaired and checked by reading, not compiled, not run on hardware
//
#include <hip/hip_runtime.h>
#include <hip/hip_bf16.h>

typedef __attribute__((ext_vector_type(16))) _Float16 v16h;
typedef __attribute__((ext_vector_type(8)))  float    v8f;

#define NQ 4
#define NL 3

__device__ inline float2 cmul(float2 a, float2 b) {
  return make_float2(a.x*b.x - a.y*b.y, a.x*b.y + a.y*b.x);
}

// ---------------------------------------------------------------------------
// Prep kernel (runs once, 1 block): builds the 96x16 f16 coefficient operand
// C such that  <Z_i>(x) = sum_j C[j,i] * prod_w phi_{j_w}(x_w),
// phi = (1, cos x, sin x), from the fixed 16x16 circuit unitary U(weights).
// Packs C into the WMMA f16 B operand in contiguous k order (k = 16h + s) in d_ws.
// ---------------------------------------------------------------------------
__global__ __launch_bounds__(256) void qlayer_prep(const float* __restrict__ wgt,
                                                   unsigned int* __restrict__ Bpack) {
  __shared__ float Ur[16][16];
  __shared__ float Ui[16][16];
  __shared__ float Amat[4][16][16];
  __shared__ float Cmat[81][4];
  const int t = threadIdx.x;

  // ---- Step 1: threads 0..15 each simulate one basis column of U ----------
  if (t < 16) {
    float2 st[16];
#pragma unroll
    for (int a = 0; a < 16; ++a) st[a] = make_float2(a == t ? 1.f : 0.f, 0.f);
#pragma unroll
    for (int l = 0; l < NL; ++l) {
#pragma unroll
      for (int w = 0; w < NQ; ++w) {
        const float phi = wgt[(l*NQ + w)*3 + 0];
        const float th  = wgt[(l*NQ + w)*3 + 1];
        const float om  = wgt[(l*NQ + w)*3 + 2];
        // Rot(phi,theta,omega) = RZ(omega) RY(theta) RZ(phi)
        const float ct = cosf(0.5f*th), sn = sinf(0.5f*th);
        const float ap = 0.5f*(phi+om), am = 0.5f*(phi-om);
        const float cap = cosf(ap), sap = sinf(ap);
        const float cam = cosf(am), sam = sinf(am);
        const float2 u00 = make_float2( cap*ct, -sap*ct);   //  e^{-i(p+o)/2} c
        const float2 u01 = make_float2(-cam*sn, -sam*sn);   // -e^{ i(p-o)/2} s
        const float2 u10 = make_float2( cam*sn, -sam*sn);   //  e^{-i(p-o)/2} s
        const float2 u11 = make_float2( cap*ct,  sap*ct);   //  e^{ i(p+o)/2} c
        const int mask = 1 << (3 - w);                       // wire w <-> bit 3-w
#pragma unroll
        for (int a = 0; a < 16; ++a) {
          if (!(a & mask)) {
            const float2 sa = st[a], sb = st[a | mask];
            float2 na = cmul(u00, sa), nb = cmul(u01, sb);
            float2 nc = cmul(u10, sa), nd = cmul(u11, sb);
            st[a]        = make_float2(na.x + nb.x, na.y + nb.y);
            st[a | mask] = make_float2(nc.x + nd.x, nc.y + nd.y);
          }
        }
      }
      const int r = (l % (NQ - 1)) + 1;   // StronglyEntanglingLayers ranges
#pragma unroll
      for (int i = 0; i < NQ; ++i) {
        const int tq = (i + r) % NQ;
        const int cm = 1 << (3 - i), tm = 1 << (3 - tq);
#pragma unroll
        for (int a = 0; a < 16; ++a) {
          if ((a & cm) && !(a & tm)) {           // control=1, target=0: swap
            const float2 tmp = st[a]; st[a] = st[a | tm]; st[a | tm] = tmp;
          }
        }
      }
    }
#pragma unroll
    for (int a = 0; a < 16; ++a) { Ur[a][t] = st[a].x; Ui[a][t] = st[a].y; }
  }
  __syncthreads();

  // ---- Step 2: A_i[a,b] = Re[(U^dag D_i U)]_{ab} --------------------------
  {
    const int a = t >> 4, b = t & 15;
#pragma unroll
    for (int i = 0; i < 4; ++i) {
      float s = 0.f;
#pragma unroll
      for (int k = 0; k < 16; ++k) {
        const float sg = ((k >> (3 - i)) & 1) ? -1.f : 1.f;
        s += sg * (Ur[k][a]*Ur[k][b] + Ui[k][a]*Ui[k][b]);
      }
      Amat[i][a][b] = s;
    }
  }
  __syncthreads();

  // ---- Step 3: half-angle basis change -> C[81][4] ------------------------
  if (t < 81) {
    const int jw[4] = { t/27, (t/9)%3, (t/3)%3, t%3 };
    float acc[4] = {0.f, 0.f, 0.f, 0.f};
    for (int a = 0; a < 16; ++a) {
      for (int b = 0; b < 16; ++b) {
        float wq = 1.f;
#pragma unroll
        for (int w = 0; w < 4; ++w) {
          const int aw = (a >> (3 - w)) & 1;
          const int bw = (b >> (3 - w)) & 1;
          const int jq = jw[w];
          float tv;
          if (aw == bw) tv = (jq == 0) ? 0.5f : ((jq == 1) ? (aw ? -0.5f : 0.5f) : 0.f);
          else          tv = (jq == 2) ? 0.5f : 0.f;
          wq *= tv;
        }
        if (wq != 0.f) {
#pragma unroll
          for (int i = 0; i < 4; ++i) acc[i] += wq * Amat[i][a][b];
        }
      }
    }
#pragma unroll
    for (int i = 0; i < 4; ++i) Cmat[t][i] = acc[i];
  }
  __syncthreads();

  // ---- Step 4: pack into f16 B-operand layout (3 WMMAs x 8 regs x 32 lanes)
  // B (32x16 per WMMA): lanes 0-15 hold K=0..15 (2 per VGPR), lanes 16-31 hold
  // K=16..31; column N = lane%16 (only N<4 nonzero). Stored [w][lane][reg].
  for (int u = t; u < 768; u += 256) {
    const int w = u >> 8, rem = u & 255;
    const int lane = rem >> 3, reg = rem & 7;
    const int col = lane & 15, h = lane >> 4;
    const int k0 = 32*w + 16*h + 2*reg;
    const int k1 = k0 + 1;
    const float v0 = (k0 < 81 && col < 4) ? Cmat[k0][col] : 0.f;
    const float v1 = (k1 < 81 && col < 4) ? Cmat[k1][col] : 0.f;
    const unsigned short h0 = __builtin_bit_cast(unsigned short, (_Float16)v0);
    const unsigned short h1 = __builtin_bit_cast(unsigned short, (_Float16)v1);
    Bpack[u] = (unsigned int)h0 | ((unsigned int)h1 << 16);
  }
}

// ---------------------------------------------------------------------------
// Main kernel: one wave per 16 items. Builds the f16 A-operand (feature rows)
// in contiguous k order (k = 16h + e), runs 3 WMMAs, writes coalesced via LDS.
// ---------------------------------------------------------------------------
template <int H>
__device__ inline void build_feats(v16h (&amat)[3],
                                   const float (&p01)[9], const float (&p23)[9]) {
#pragma unroll
  for (int w = 0; w < 3; ++w) {
    v16h av;
#pragma unroll
    for (int e = 0; e < 16; ++e) {
      // A 16x32 f16 layout: half-lane H covers K = 16H+0..15 (regs 0-7),
      // within K-chunk 32w (matches B pack).
      const int k = 32*w + 16*H + e;
      float f = 0.f;
      if (k < 81) f = p01[k/9] * p23[k%9];
      av[e] = (_Float16)f;
    }
    amat[w] = av;
  }
}

__global__ __launch_bounds__(256) void qlayer_main(const float* __restrict__ x,
                                                   const unsigned int* __restrict__ Bpack,
                                                   float* __restrict__ out,
                                                   const int ntiles, const int nitems) {
  __shared__ float stage[8][64];
  const int lane = threadIdx.x & 31;
  const int wid  = threadIdx.x >> 5;
  const int tile = blockIdx.x * 8 + wid;
  if (tile >= ntiles) return;          // wave-uniform: EXEC stays all-ones

  // Shared coefficient operand, resident in VGPRs (six b128 loads per lane).
  v16h bmat[3];
#pragma unroll
  for (int w = 0; w < 3; ++w)
    bmat[w] = *reinterpret_cast<const v16h*>(Bpack + w*256 + lane*8);

  int item = tile*16 + (lane & 15);
  if (item >= nitems) item = nitems - 1;
  const float4 xi = reinterpret_cast<const float4*>(x)[item];

  float s0, c0, s1, c1, s2, c2, s3, c3;
  __sincosf(xi.x, &s0, &c0);
  __sincosf(xi.y, &s1, &c1);
  __sincosf(xi.z, &s2, &c2);
  __sincosf(xi.w, &s3, &c3);

  const float p01[9] = {1.f, c1, s1, c0, c0*c1, c0*s1, s0, s0*c1, s0*s1};
  const float p23[9] = {1.f, c3, s3, c2, c2*c3, c2*s3, s2, s2*c3, s2*s3};

  v16h amat[3];
  if (lane < 16) build_feats<0>(amat, p01, p23);
  else           build_feats<1>(amat, p01, p23);

  v8f acc = {};   // D rows = items in tile, cols = output qubit (N<4 used)
#pragma unroll
  for (int w = 0; w < 3; ++w)
    acc = __builtin_amdgcn_wmma_f32_16x16x32_f16(false, amat[w], false, bmat[w],
                                                 (short)0, acc, false, false);

  // D layout: VGPR k, lane halves -> (M = k + 8h, N = lane%16). Stage the
  // 4 useful columns to LDS, then emit one coalesced 256B float2 store/wave.
  const int n = lane & 15, h = lane >> 4;
  if (n < 4) {
#pragma unroll
    for (int k = 0; k < 8; ++k) stage[wid][(k + 8*h)*4 + n] = acc[k];
  }
  asm volatile("s_wait_dscnt 0" ::: "memory");   // wave-local LDS visibility
  const float2 o = *reinterpret_cast<const float2*>(&stage[wid][lane*2]);
  const long ofs = (long)tile*32 + lane;
  if (ofs < (long)nitems*2)
    reinterpret_cast<float2*>(out)[ofs] = o;
}

extern "C" void kernel_launch(void* const* d_in, const int* in_sizes, int n_in,
                              void* d_out, int out_size, void* d_ws, size_t ws_size,
                              hipStream_t stream) {
  (void)n_in; (void)out_size; (void)ws_size;
  const float* x   = (const float*)d_in[0];        // (B, 4) float32
  const float* wgt = (const float*)d_in[1];        // (3, 4, 3) float32
  float* out = (float*)d_out;                      // (B, 4) float32
  unsigned int* Bpack = (unsigned int*)d_ws;       // 768 dwords = 3 KB

  const int nitems  = in_sizes[0] / NQ;
  const int ntiles  = (nitems + 15) / 16;
  const int nblocks = (ntiles + 7) / 8;

  hipLaunchKernelGGL(qlayer_prep, dim3(1), dim3(256), 0, stream, wgt, Bpack);
  hipLaunchKernelGGL(qlayer_main, dim3(nblocks), dim3(256), 0, stream,
                     x, Bpack, out, ntiles, nitems);
}
